// PillarFeatureNet_trt_88845693485726
// MI455X (gfx1250) — compile-verified, matrix operand order checked by reading, not run on hardware
//
#include <hip/hip_runtime.h>

// PillarFeatureNet fused: features -> Linear(9,32) via WMMA -> BN -> ReLU -> max -> concat
// One workgroup (7 wave32s) per pillar. Output-bandwidth bound (~307 MB write).

#define PILLARS 12000
#define NPTS    100
#define CIN     9
#define COUT    32
#define BN_EPS  1e-3f
#define NTILES  7            // ceil(100/16)
#define THREADS 224          // 7 waves * 32 lanes

typedef __attribute__((ext_vector_type(16))) _Float16 v16h;
typedef __attribute__((ext_vector_type(8)))  float    v8f;

__global__ __launch_bounds__(THREADS)
void pfn_fused_wmma(const float* __restrict__ px, const float* __restrict__ py,
                    const float* __restrict__ pz, const float* __restrict__ pi,
                    const int*   __restrict__ nvox,
                    const float* __restrict__ xs, const float* __restrict__ ys,
                    const float* __restrict__ mk,
                    const float* __restrict__ W,
                    const float* __restrict__ gam, const float* __restrict__ bet,
                    const float* __restrict__ bmean, const float* __restrict__ bvar,
                    float* __restrict__ out)
{
    __shared__ float redx[128], redy[128], redz[128];
    __shared__ float sxout[112 * COUT];          // BN+ReLU'd x, rows padded to 112
    __shared__ float smaxA[NTILES * 32];         // per-wave partial maxima, ch 0..15 (2 slots/ch)
    __shared__ float smaxB[NTILES * 32];         // per-wave partial maxima, ch 16..31
    __shared__ float sfinal[COUT];               // per-pillar channel max

    const int  p    = blockIdx.x;
    const int  t    = threadIdx.x;
    const int  wv   = t >> 5;                    // wave id = M-tile id, 0..6
    const int  lane = t & 31;
    const long base = (long)p * NPTS;

    // ---- points_mean: sum over all N slots / num_voxels (reference semantics) ----
    if (t < 128) {
        float vx = 0.f, vy = 0.f, vz = 0.f;
        if (t < NPTS) { vx = px[base + t]; vy = py[base + t]; vz = pz[base + t]; }
        redx[t] = vx; redy[t] = vy; redz[t] = vz;
    }
    __syncthreads();
    #pragma unroll
    for (int off = 64; off > 0; off >>= 1) {
        if (t < off) {
            redx[t] += redx[t + off];
            redy[t] += redy[t + off];
            redz[t] += redz[t + off];
        }
        __syncthreads();
    }
    const float inv_nv = 1.0f / (float)nvox[p];
    const float mx = redx[0] * inv_nv;
    const float my = redy[0] * inv_nv;
    const float mz = redz[0] * inv_nv;

    // ---- A fragment: 16x32 f16, row = wv*16 + (lane&15); K=0..8 valid, rest zero ----
    const int rrow = lane & 15;
    const int r    = wv * 16 + rrow;             // point index packed by this lane
    float m = 0.f, x = 0.f, y = 0.f, z = 0.f, ii = 0.f, xsv = 0.f, ysv = 0.f;
    if (r < NPTS) {
        m   = mk[base + r];
        x   = px[base + r];  y   = py[base + r];  z = pz[base + r];
        ii  = pi[base + r];  xsv = xs[base + r];  ysv = ys[base + r];
    }
    float f[CIN];
    f[0] = x * m;          f[1] = y * m;          f[2] = z * m;  f[3] = ii * m;
    f[4] = (x - mx) * m;   f[5] = (y - my) * m;   f[6] = (z - mz) * m;
    f[7] = (x - xsv) * m;  f[8] = (y - ysv) * m;

    v16h a;
    #pragma unroll
    for (int k = 0; k < 16; ++k) a[k] = (_Float16)0.f;
    if (lane < 16) {                              // K = 0..7 in elements 0..7
        #pragma unroll
        for (int k = 0; k < 8; ++k) a[k] = (_Float16)f[k];
    } else {                                      // K = 8 in element 0
        a[0] = (_Float16)f[8];
    }

    // ---- B fragments: 32x16 f16, B[k][n] = W[n*9+k]; col = lane&15 (+16 for b1) ----
    const int n0 = lane & 15;
    v16h b0, b1;
    #pragma unroll
    for (int k = 0; k < 16; ++k) { b0[k] = (_Float16)0.f; b1[k] = (_Float16)0.f; }
    if (lane < 16) {
        #pragma unroll
        for (int k = 0; k < 8; ++k) {
            b0[k] = (_Float16)W[n0 * CIN + k];
            b1[k] = (_Float16)W[(n0 + 16) * CIN + k];
        }
    } else {
        b0[0] = (_Float16)W[n0 * CIN + 8];
        b1[0] = (_Float16)W[(n0 + 16) * CIN + 8];
    }

    // ---- WMMA: D = A x B (f32 accumulate), two 16-wide column tiles ----
    v8f c0, c1;
    #pragma unroll
    for (int v = 0; v < 8; ++v) { c0[v] = 0.f; c1[v] = 0.f; }
    c0 = __builtin_amdgcn_wmma_f32_16x16x32_f16(false, a, false, b0, (short)0, c0, false, false);
    c1 = __builtin_amdgcn_wmma_f32_16x16x32_f16(false, a, false, b1, (short)0, c1, false, false);

    // ---- BN (eval) + ReLU + partial max over valid rows ----
    const int ch0 = lane & 15, ch1 = ch0 + 16;
    const float s0  = gam[ch0] * rsqrtf(bvar[ch0] + BN_EPS);
    const float s1  = gam[ch1] * rsqrtf(bvar[ch1] + BN_EPS);
    const float sh0 = bet[ch0] - bmean[ch0] * s0;
    const float sh1 = bet[ch1] - bmean[ch1] * s1;

    const int rbase = wv * 16 + ((lane >> 4) << 3);   // this lane's rows: rbase..rbase+7
    float pm0 = 0.f, pm1 = 0.f;                        // relu >= 0 so 0-init is safe
    #pragma unroll
    for (int v = 0; v < 8; ++v) {
        const int gr = rbase + v;
        const float y0 = fmaxf(c0[v] * s0 + sh0, 0.f);
        const float y1 = fmaxf(c1[v] * s1 + sh1, 0.f);
        sxout[gr * COUT + ch0] = y0;
        sxout[gr * COUT + ch1] = y1;
        if (gr < NPTS) { pm0 = fmaxf(pm0, y0); pm1 = fmaxf(pm1, y1); }
    }
    smaxA[wv * 32 + lane] = pm0;
    smaxB[wv * 32 + lane] = pm1;
    __syncthreads();

    // ---- final per-channel max across 7 tiles (2 lane-slots per channel) ----
    if (t < COUT) {
        float fm = 0.f;
        if (t < 16) {
            #pragma unroll
            for (int w = 0; w < NTILES; ++w)
                fm = fmaxf(fm, fmaxf(smaxA[w * 32 + t], smaxA[w * 32 + t + 16]));
        } else {
            const int cc = t - 16;
            #pragma unroll
            for (int w = 0; w < NTILES; ++w)
                fm = fmaxf(fm, fmaxf(smaxB[w * 32 + cc], smaxB[w * 32 + cc + 16]));
        }
        sfinal[t] = fm;
    }
    __syncthreads();

    // ---- streaming write: [N,64] = concat(x, broadcast(max)); float4 coalesced ----
    float* outp = out + base * 64;
    for (int idx = t; idx < NPTS * 16; idx += THREADS) {
        const int row = idx >> 4;
        const int q   = idx & 15;
        float4 v4;
        if (q < 8) {
            const int o = row * COUT + q * 4;
            v4 = make_float4(sxout[o], sxout[o + 1], sxout[o + 2], sxout[o + 3]);
        } else {
            const int o = (q - 8) * 4;
            v4 = make_float4(sfinal[o], sfinal[o + 1], sfinal[o + 2], sfinal[o + 3]);
        }
        *(float4*)(outp + row * 64 + q * 4) = v4;
    }
}

extern "C" void kernel_launch(void* const* d_in, const int* in_sizes, int n_in,
                              void* d_out, int out_size, void* d_ws, size_t ws_size,
                              hipStream_t stream) {
    (void)in_sizes; (void)n_in; (void)out_size; (void)d_ws; (void)ws_size;
    const float* px    = (const float*)d_in[0];
    const float* py    = (const float*)d_in[1];
    const float* pz    = (const float*)d_in[2];
    const float* pi    = (const float*)d_in[3];
    const int*   nvox  = (const int*)  d_in[4];
    const float* xs    = (const float*)d_in[5];
    const float* ys    = (const float*)d_in[6];
    const float* mk    = (const float*)d_in[7];
    const float* W     = (const float*)d_in[8];
    const float* gam   = (const float*)d_in[9];
    const float* bet   = (const float*)d_in[10];
    const float* bmean = (const float*)d_in[11];
    const float* bvar  = (const float*)d_in[12];
    float* out = (float*)d_out;

    pfn_fused_wmma<<<PILLARS, THREADS, 0, stream>>>(
        px, py, pz, pi, nvox, xs, ys, mk, W, gam, bet, bmean, bvar, out);
}
